// GATv2_515396076081
// MI455X (gfx1250) — matrix operand order repaired and checked by reading, not compiled, not run on hardware
//
#include <hip/hip_runtime.h>
#include <hip/hip_bf16.h>
#include <cstdint>

// ---------------------------------------------------------------------------
// GATv2 (2 layers) + output linear + log_softmax for MI455X (gfx1250, wave32)
// GEMMs run on v_wmma_f32_16x16x32_f16; edge softmax/aggregation is
// wave-per-edge with float4 lanes (coalesced 512B per node row).
// ---------------------------------------------------------------------------

typedef __attribute__((ext_vector_type(16))) _Float16 v16h;
typedef __attribute__((ext_vector_type(8)))  _Float16 v8h;
typedef __attribute__((ext_vector_type(8)))  float    v8f;

#define NEG_SLOPE 0.2f
#define SOFT_EPS  1e-16f
#define DIM 128

// ---------------- conversion helpers ----------------

__global__ void f32_to_f16_kernel(const float* __restrict__ in,
                                  _Float16* __restrict__ out, int n) {
    int i = blockIdx.x * blockDim.x + threadIdx.x;
    if (i < n) out[i] = (_Float16)in[i];
}

// Wt[h*K + k] = W[k*H + h], converted to f16  (K = H = 128)
__global__ void transpose_f16_kernel(const float* __restrict__ W,
                                     _Float16* __restrict__ Wt, int K, int H) {
    int i = blockIdx.x * blockDim.x + threadIdx.x;
    if (i < K * H) {
        int k = i / H, h = i % H;
        Wt[h * K + k] = (_Float16)W[i];
    }
}

// ---------------- WMMA GEMM: out[N,128] = A[N,128] @ W + bias ----------------
// A: row-major f16. Wt: pre-transposed f16, Wt[h][k] = W[k][h].
// One wave per 16x16 output tile; 8 waves/block cover all 128 output cols.
// blockIdx.y selects (Wt,bias,out) pair so xl and xr run in one launch.

__global__ void wmma_gemm128_kernel(const _Float16* __restrict__ A,
                                    const _Float16* __restrict__ Wt0,
                                    const float* __restrict__ bias0,
                                    float* __restrict__ out0,
                                    const _Float16* __restrict__ Wt1,
                                    const float* __restrict__ bias1,
                                    float* __restrict__ out1,
                                    int N) {
    const int K = DIM, H = DIM;
    const _Float16* Wt  = (blockIdx.y == 0) ? Wt0  : Wt1;
    const float*   bias = (blockIdx.y == 0) ? bias0 : bias1;
    float*         out  = (blockIdx.y == 0) ? out0  : out1;

    int lane = threadIdx.x & 31;
    int wv   = threadIdx.x >> 5;         // 0..7 -> output column tile
    int row0 = blockIdx.x * 16;
    int col0 = wv * 16;
    if (row0 >= N) return;

    int m  = lane & 15;                  // A row within tile
    int n  = lane & 15;                  // B/D col within tile
    int hi = lane >> 4;                  // half-wave select
    int ka = hi * 16;                    // A K-base   (contiguous, matches B)
    int kb = hi * 16;                    // B K-base   (contiguous, matches A)

    const _Float16* arow = A  + (size_t)(row0 + m) * K;
    const _Float16* brow = Wt + (size_t)(col0 + n) * K;

    v8f c = {};
#pragma unroll
    for (int k = 0; k < K; k += 32) {
        v8h alo = *(const v8h*)(arow + k + ka);        // K = ka+0..7
        v8h ahi = *(const v8h*)(arow + k + ka + 8);    // K = ka+8..15
        v16h a;
#pragma unroll
        for (int i = 0; i < 8; ++i) { a[i] = alo[i]; a[i + 8] = ahi[i]; }
        v16h b = *(const v16h*)(brow + k + kb);        // K = kb+0..15, contiguous
        c = __builtin_amdgcn_wmma_f32_16x16x32_f16(
                /*neg_a=*/false, a, /*neg_b=*/false, b,
                /*c_mod=*/(short)0, c, /*reuse_a=*/false, /*reuse_b=*/false);
    }

    float bcol = bias[col0 + n];
#pragma unroll
    for (int v = 0; v < 8; ++v) {
        int mo = v + hi * 8;                           // ISA f32 C/D layout
        out[(size_t)(row0 + mo) * H + col0 + n] = c[v] + bcol;
    }
}

// ---------------- per-call buffer init ----------------

__global__ void init_buffers_kernel(float* __restrict__ mbuf,
                                    float* __restrict__ denom,
                                    float* __restrict__ accum,
                                    int N, int total) {
    int i = blockIdx.x * blockDim.x + threadIdx.x;
    if (i < total) accum[i] = 0.0f;
    if (i < N) { mbuf[i] = -3.0e38f; denom[i] = 0.0f; }
}

// ---------------- edge pass A: score + segment max ----------------

__device__ inline void atomicMaxF(float* addr, float val) {
    unsigned int* ua = (unsigned int*)addr;
    unsigned int old = *ua;
    while (__uint_as_float(old) < val) {
        unsigned int assumed = old;
        old = atomicCAS(ua, assumed, __float_as_uint(val));
        if (old == assumed) break;
    }
}

__global__ void edge_score_kernel(const float* __restrict__ xl,
                                  const float* __restrict__ xr,
                                  const float* __restrict__ att,
                                  const int* __restrict__ src,
                                  const int* __restrict__ dst,
                                  float* __restrict__ ebuf,
                                  float* __restrict__ mbuf, int E) {
    int edge = blockIdx.x * (blockDim.x >> 5) + (threadIdx.x >> 5);
    int lane = threadIdx.x & 31;
    if (edge >= E) return;
    int s = src[edge], d = dst[edge];

    float4 a = *(const float4*)(xl + (size_t)s * DIM + lane * 4);
    float4 b = *(const float4*)(xr + (size_t)d * DIM + lane * 4);
    float4 t = *(const float4*)(att + lane * 4);

    float p = 0.0f, v;
    v = a.x + b.x; v = v > 0.0f ? v : NEG_SLOPE * v; p += v * t.x;
    v = a.y + b.y; v = v > 0.0f ? v : NEG_SLOPE * v; p += v * t.y;
    v = a.z + b.z; v = v > 0.0f ? v : NEG_SLOPE * v; p += v * t.z;
    v = a.w + b.w; v = v > 0.0f ? v : NEG_SLOPE * v; p += v * t.w;

#pragma unroll
    for (int off = 16; off > 0; off >>= 1) p += __shfl_xor(p, off, 32);

    if (lane == 0) {
        ebuf[edge] = p;
        atomicMaxF(mbuf + d, p);
    }
}

// ---------------- edge pass B: exp + segment sum ----------------

__global__ void edge_exp_kernel(const int* __restrict__ dst,
                                const float* __restrict__ mbuf,
                                float* __restrict__ ebuf,   // in: e, out: w (in place)
                                float* __restrict__ denom, int E) {
    int i = blockIdx.x * blockDim.x + threadIdx.x;
    if (i >= E) return;
    int d = dst[i];
    float w = __expf(ebuf[i] - mbuf[d]);
    ebuf[i] = w;
    atomicAdd(denom + d, w);
}

// ---------------- edge pass C: alpha-weighted scatter-add ----------------

__global__ void edge_aggregate_kernel(const float* __restrict__ xl,
                                      const float* __restrict__ wbuf,
                                      const float* __restrict__ denom,
                                      const int* __restrict__ src,
                                      const int* __restrict__ dst,
                                      float* __restrict__ accum, int E) {
    int edge = blockIdx.x * (blockDim.x >> 5) + (threadIdx.x >> 5);
    int lane = threadIdx.x & 31;
    if (edge >= E) return;
    int s = src[edge], d = dst[edge];
    float alpha = wbuf[edge] / (denom[d] + SOFT_EPS);

    float4 a = *(const float4*)(xl + (size_t)s * DIM + lane * 4);
    float* o = accum + (size_t)d * DIM + lane * 4;
    atomicAdd(o + 0, a.x * alpha);
    atomicAdd(o + 1, a.y * alpha);
    atomicAdd(o + 2, a.z * alpha);
    atomicAdd(o + 3, a.w * alpha);
}

// ---------------- pass D: bias + ReLU, emit f32 + f16 ----------------

__global__ void finalize_node_kernel(const float* __restrict__ accum,
                                     const float* __restrict__ bias,
                                     float* __restrict__ hf32,
                                     _Float16* __restrict__ hf16, int total) {
    int i = blockIdx.x * blockDim.x + threadIdx.x;
    if (i >= total) return;
    float v = accum[i] + bias[i & (DIM - 1)];
    v = v > 0.0f ? v : 0.0f;
    hf32[i] = v;
    hf16[i] = (_Float16)v;
}

// ---------------- output linear (128->10) + log_softmax ----------------

__global__ void out_logsoftmax_kernel(const float* __restrict__ h,
                                      const float* __restrict__ Wo,
                                      const float* __restrict__ bo,
                                      float* __restrict__ out, int N) {
    int nid = blockIdx.x * blockDim.x + threadIdx.x;
    if (nid >= N) return;
    const int C = 10;
    float logits[C];
#pragma unroll
    for (int c = 0; c < C; ++c) logits[c] = bo[c];
    const float* hr = h + (size_t)nid * DIM;
    for (int k = 0; k < DIM; ++k) {
        float hv = hr[k];
        const float* wr = Wo + k * C;
#pragma unroll
        for (int c = 0; c < C; ++c) logits[c] += hv * wr[c];
    }
    float mx = logits[0];
#pragma unroll
    for (int c = 1; c < C; ++c) mx = fmaxf(mx, logits[c]);
    float se = 0.0f;
#pragma unroll
    for (int c = 0; c < C; ++c) se += __expf(logits[c] - mx);
    float lse = mx + __logf(se);
#pragma unroll
    for (int c = 0; c < C; ++c) out[(size_t)nid * C + c] = logits[c] - lse;
}

// ---------------------------------------------------------------------------
// host-side launcher
// ---------------------------------------------------------------------------

static inline size_t align256(size_t x) { return (x + 255) & ~size_t(255); }

extern "C" void kernel_launch(void* const* d_in, const int* in_sizes, int n_in,
                              void* d_out, int out_size, void* d_ws, size_t ws_size,
                              hipStream_t stream) {
    const float* x   = (const float*)d_in[0];
    const int*   ei  = (const int*)d_in[1];
    const float* W1l = (const float*)d_in[2];
    const float* b1l = (const float*)d_in[3];
    const float* W1r = (const float*)d_in[4];
    const float* b1r = (const float*)d_in[5];
    const float* att1  = (const float*)d_in[6];
    const float* bias1 = (const float*)d_in[7];
    const float* W2l = (const float*)d_in[8];
    const float* b2l = (const float*)d_in[9];
    const float* W2r = (const float*)d_in[10];
    const float* b2r = (const float*)d_in[11];
    const float* att2  = (const float*)d_in[12];
    const float* bias2 = (const float*)d_in[13];
    const float* Wo = (const float*)d_in[14];
    const float* bo = (const float*)d_in[15];
    float* out = (float*)d_out;

    const int N = in_sizes[0] / DIM;      // 40000
    const int E = in_sizes[1] / 2;        // 640000
    const int* src = ei;
    const int* dst = ei + E;
    const int total = N * DIM;

    // ---- workspace layout ----
    uint8_t* ws = (uint8_t*)d_ws;
    size_t off = 0;
    _Float16* Abf  = (_Float16*)(ws + off); off = align256(off + (size_t)total * sizeof(_Float16));
    float*    xl   = (float*)(ws + off);    off = align256(off + (size_t)total * sizeof(float));
    float*    xr   = (float*)(ws + off);    off = align256(off + (size_t)total * sizeof(float));
    float*    acc  = (float*)(ws + off);    off = align256(off + (size_t)total * sizeof(float));
    float*    hf32 = (float*)(ws + off);    off = align256(off + (size_t)total * sizeof(float));
    float*    ebuf = (float*)(ws + off);    off = align256(off + (size_t)E * sizeof(float));
    float*    mbuf = (float*)(ws + off);    off = align256(off + (size_t)N * sizeof(float));
    float*    denm = (float*)(ws + off);    off = align256(off + (size_t)N * sizeof(float));
    _Float16* Wtl  = (_Float16*)(ws + off); off = align256(off + (size_t)DIM * DIM * sizeof(_Float16));
    _Float16* Wtr  = (_Float16*)(ws + off); off = align256(off + (size_t)DIM * DIM * sizeof(_Float16));
    (void)ws_size; (void)n_in; (void)out_size;

    const int TB = 256;
    const int blocksTotal = (total + TB - 1) / TB;          // element-wise grids
    const int blocksW     = (DIM * DIM + TB - 1) / TB;      // weight transpose
    const int edgesPerBlk = TB / 32;                        // wave-per-edge grids
    const int blocksWaveE = (E + edgesPerBlk - 1) / edgesPerBlk;
    const int blocksThrE  = (E + TB - 1) / TB;
    dim3 gemmGrid(N / 16, 2);                               // .y: 0 = l, 1 = r

    // ===== layer 1 =====
    f32_to_f16_kernel<<<blocksTotal, TB, 0, stream>>>(x, Abf, total);
    transpose_f16_kernel<<<blocksW, TB, 0, stream>>>(W1l, Wtl, DIM, DIM);
    transpose_f16_kernel<<<blocksW, TB, 0, stream>>>(W1r, Wtr, DIM, DIM);
    wmma_gemm128_kernel<<<gemmGrid, TB, 0, stream>>>(Abf, Wtl, b1l, xl, Wtr, b1r, xr, N);
    init_buffers_kernel<<<blocksTotal, TB, 0, stream>>>(mbuf, denm, acc, N, total);
    edge_score_kernel<<<blocksWaveE, TB, 0, stream>>>(xl, xr, att1, src, dst, ebuf, mbuf, E);
    edge_exp_kernel<<<blocksThrE, TB, 0, stream>>>(dst, mbuf, ebuf, denm, E);
    edge_aggregate_kernel<<<blocksWaveE, TB, 0, stream>>>(xl, ebuf, denm, src, dst, acc, E);
    finalize_node_kernel<<<blocksTotal, TB, 0, stream>>>(acc, bias1, hf32, Abf, total);

    // ===== layer 2 (reuses all buffers; stream order gives dependencies) =====
    transpose_f16_kernel<<<blocksW, TB, 0, stream>>>(W2l, Wtl, DIM, DIM);
    transpose_f16_kernel<<<blocksW, TB, 0, stream>>>(W2r, Wtr, DIM, DIM);
    wmma_gemm128_kernel<<<gemmGrid, TB, 0, stream>>>(Abf, Wtl, b2l, xl, Wtr, b2r, xr, N);
    init_buffers_kernel<<<blocksTotal, TB, 0, stream>>>(mbuf, denm, acc, N, total);
    edge_score_kernel<<<blocksWaveE, TB, 0, stream>>>(xl, xr, att2, src, dst, ebuf, mbuf, E);
    edge_exp_kernel<<<blocksThrE, TB, 0, stream>>>(dst, mbuf, ebuf, denm, E);
    edge_aggregate_kernel<<<blocksWaveE, TB, 0, stream>>>(xl, ebuf, denm, src, dst, acc, E);
    finalize_node_kernel<<<blocksTotal, TB, 0, stream>>>(acc, bias2, hf32, Abf, total);

    // ===== output head =====
    out_logsoftmax_kernel<<<(N + TB - 1) / TB, TB, 0, stream>>>(hf32, Wo, bo, out, N);
}
